// SpikingLayer_15247133901065
// MI455X (gfx1250) — matrix operand order repaired and checked by reading, not compiled, not run on hardware
//
#include <hip/hip_runtime.h>

// ---------------------------------------------------------------------------
// SLSTM on MI455X (gfx1250): all GEMMs via v_wmma_f32_16x16x32_bf16 (4Mx4N
// tiles per wave), recurrence as a persistent kernel with W_hh staged into
// 256KB LDS/WGP by the Tensor Data Mover (tensor_load_to_lds + TENSORcnt).
// ---------------------------------------------------------------------------

typedef __bf16 bf16_t;
typedef __attribute__((ext_vector_type(16))) __bf16 v16bf;
typedef __attribute__((ext_vector_type(8)))  float  v8f;
typedef __attribute__((ext_vector_type(4)))  unsigned int u32x4;
typedef __attribute__((ext_vector_type(8)))  int  i32x8;
typedef __attribute__((ext_vector_type(4)))  int  i32x4;

#define S_LEN 2048
#define B_SZ  8
#define D_DIM 1024
#define H_DIM 1024
#define G4    4096
#define NT    16384           // B*S tokens
#define SCAN_WGS 32           // workgroups in persistent scan kernel

// ---- bf16 <-> f32 helpers (RNE) -------------------------------------------
__device__ __forceinline__ bf16_t to_bf(float f) {
    unsigned u = __float_as_uint(f);
    unsigned r = (u + 0x7FFFu + ((u >> 16) & 1u)) >> 16;
    unsigned short us = (unsigned short)r;
    bf16_t out;
    __builtin_memcpy(&out, &us, 2);
    return out;
}
__device__ __forceinline__ float from_bf(bf16_t b) {
    unsigned short us;
    __builtin_memcpy(&us, &b, 2);
    return __uint_as_float(((unsigned)us) << 16);
}
__device__ __forceinline__ float sigmoidf_(float x) {
    return 1.0f / (1.0f + __expf(-x));
}
// LDS byte offset of a generic pointer into shared memory.
__device__ __forceinline__ unsigned lds_offset_of(void* p) {
    return (unsigned)(unsigned long long)(__attribute__((address_space(3))) unsigned char*)p;
}

// ---- WMMA fragment loaders (row-major bf16 sources, 16B-aligned) ----------
// A (16x32): lane holds row M=lane&15, 16 contiguous K starting at
//            k0 + (lane>=16 ? 16 : 0).
__device__ __forceinline__ v16bf load_a_frag(const bf16_t* __restrict__ base,
                                             int row0, int rowStride, int k0, int lane) {
    const bf16_t* p = base + (size_t)(row0 + (lane & 15)) * rowStride
                      + k0 + ((lane >> 4) << 4);
    union { v16bf v; uint4 q[2]; } u;
    u.q[0] = *(const uint4*)(p);        // 8 bf16
    u.q[1] = *(const uint4*)(p + 8);    // 8 bf16, K+8
    return u.v;
}
// B (32x16): B[k][n] = W[col0+n][k]; lane n holds 16 contiguous K starting at
//            k0 + (lane>=16 ? 16 : 0).
__device__ __forceinline__ v16bf load_b_frag(const bf16_t* __restrict__ base,
                                             int col0, int rowStride, int k0, int lane) {
    const bf16_t* p = base + (size_t)(col0 + (lane & 15)) * rowStride
                      + k0 + ((lane >> 4) << 4);
    union { v16bf v; uint4 q[2]; } u;
    u.q[0] = *(const uint4*)(p);
    u.q[1] = *(const uint4*)(p + 8);
    return u.v;
}
#define WMMA_BF16(a, b, c) \
    __builtin_amdgcn_wmma_f32_16x16x32_bf16(false, (a), false, (b), (short)0, (c), false, false)

// ---------------------------------------------------------------------------
// Conversion / init kernels
// ---------------------------------------------------------------------------
__global__ void conv_f2b(const float* __restrict__ src, bf16_t* __restrict__ dst, int n) {
    int i = blockIdx.x * blockDim.x + threadIdx.x;
    if (i < n) dst[i] = to_bf(src[i]);
}
// Permute gate rows: dst row p = h*4 + gate  <-  src row gate*1024 + h
__global__ void conv_w_perm(const float* __restrict__ W, bf16_t* __restrict__ Wp, int cols) {
    int i = blockIdx.x * blockDim.x + threadIdx.x;
    int n = G4 * cols;
    if (i >= n) return;
    int p = i / cols, c = i - p * cols;
    int g = p & 3, h = p >> 2;
    Wp[i] = to_bf(W[(size_t)(g * H_DIM + h) * cols + c]);
}
__global__ void conv_bias_perm(const float* __restrict__ bi, const float* __restrict__ bh,
                               float* __restrict__ bg) {
    int p = blockIdx.x * blockDim.x + threadIdx.x;
    if (p >= G4) return;
    int g = p & 3, h = p >> 2;
    bg[p] = bi[g * H_DIM + h] + bh[g * H_DIM + h];
}
__global__ void init_state(unsigned short* __restrict__ membuf, unsigned int* __restrict__ bar) {
    int i = blockIdx.x * blockDim.x + threadIdx.x;
    if (i < 2 * 16 * H_DIM) membuf[i] = 0;   // both mem ping-pong buffers, rows 8..15 stay 0
    if (i == 0) *bar = 0u;
}

// ---------------------------------------------------------------------------
// GEMM 1: xg[t][p] = x_bf16[t][:] . Wih_perm[p][:] + biasg[p]   (bf16 out)
// Wave = 4 M-tiles x 4 N-tiles (16 wmma per K-step, 1 b128 load per wmma).
// grid (32, 64), block 256.
// ---------------------------------------------------------------------------
__global__ void __launch_bounds__(256) gemm_xg(const bf16_t* __restrict__ xb,
                                               const bf16_t* __restrict__ Wp,
                                               const float* __restrict__ biasg,
                                               bf16_t* __restrict__ xg) {
    int lane = threadIdx.x & 31;
    int wave = threadIdx.x >> 5;
    int mbase = (blockIdx.x * 8 + wave) * 4;     // first of 4 M-tiles
    int nt0   = blockIdx.y * 4;                  // first of 4 N-tiles
    v8f c[16] = {};
    for (int k0 = 0; k0 < D_DIM; k0 += 32) {
        v16bf a[4];
        #pragma unroll
        for (int i = 0; i < 4; ++i)
            a[i] = load_a_frag(xb, (mbase + i) * 16, D_DIM, k0, lane);
        #pragma unroll
        for (int j = 0; j < 4; ++j) {
            v16bf b = load_b_frag(Wp, (nt0 + j) * 16, D_DIM, k0, lane);
            #pragma unroll
            for (int i = 0; i < 4; ++i)
                c[i * 4 + j] = WMMA_BF16(a[i], b, c[i * 4 + j]);
        }
    }
    int n = lane & 15, mhi = (lane >> 4) * 8;
    #pragma unroll
    for (int j = 0; j < 4; ++j) {
        int col = (nt0 + j) * 16 + n;
        float bv = biasg[col];
        #pragma unroll
        for (int i = 0; i < 4; ++i) {
            int trow = (mbase + i) * 16 + mhi;
            #pragma unroll
            for (int r = 0; r < 8; ++r)
                xg[(size_t)(trow + r) * G4 + col] = to_bf(c[i * 4 + j][r] + bv);
        }
    }
}

// ---------------------------------------------------------------------------
// Persistent SLSTM scan.  32 WGs x 8 waves; wave nt owns gate columns
// [nt*16, nt*16+16) == (h in [nt*4,nt*4+4)) x 4 gates, batches 0..7.
// W_hh tile (128x1024 bf16 = 256KB) staged to LDS once via the TDM.
// One global barrier per timestep; mem state double-buffered.
// ---------------------------------------------------------------------------
__global__ void __launch_bounds__(256) slstm_scan(
        const bf16_t* __restrict__ Whp,      // [4096][1024] permuted bf16
        const bf16_t* __restrict__ xg,       // [NT][4096] bf16
        const float*  __restrict__ thr_p,
        float* __restrict__ syns,            // [B][S][H] fp32
        float* __restrict__ mems,            // [B][S][H] fp32
        bf16_t* __restrict__ spks,           // [B][S][H] bf16
        bf16_t* __restrict__ membuf,         // [2][16][1024] bf16
        unsigned int* __restrict__ bar) {
    extern __shared__ unsigned char smem[];
    bf16_t* sW = (bf16_t*)smem;                          // 128 x 1024 bf16 (256KB)
    float*  sG = (float*)(smem + 128 * 1024 * 2);        // 8 waves x (8x16) floats

    int lane = threadIdx.x & 31;
    int wave = threadIdx.x >> 5;
    int nt   = blockIdx.x * 8 + wave;                    // 0..255

    // --- TDM: stage this WG's 128x1024 bf16 W_hh slab into LDS -------------
    if (threadIdx.x < 32) {  // wave 0 issues one tensor_load_to_lds
        unsigned long long ga =
            (unsigned long long)(Whp + (size_t)blockIdx.x * 128 * H_DIM);
        u32x4 g0 = {
            1u,                                     // [1:0] count=1 (user D#)
            lds_offset_of(sW),                      // [63:32] lds_addr
            (unsigned)(ga & 0xFFFFFFFFull),         // [95:64] global_addr lo
            (unsigned)((ga >> 32) & 0x01FFFFFFull)  // [120:96] global_addr hi
                | (2u << 30)                        // [127:126] type=2 (image)
        };
        i32x8 g1 = {
            (int)0x00010000u,        // workgroup_mask=0, data_size=1 (2B)
            (int)(1024u << 16),      // tensor_dim0 = 1024 (low 16 bits)
            (int)(128u << 16),       // tensor_dim0 hi=0 | tensor_dim1 = 128
            (int)(1024u << 16),      // tensor_dim1 hi=0 | tile_dim0 = 1024
            (int)128u,               // tile_dim1 = 128, tile_dim2 = 0
            (int)1024u,              // tensor_dim0_stride = 1024
            0, 0                     // stride hi / tensor_dim1_stride = 0
        };
        i32x4 gz4 = { 0, 0, 0, 0 };
        i32x8 gz8 = { 0, 0, 0, 0, 0, 0, 0, 0 };
        __builtin_amdgcn_tensor_load_to_lds(g0, g1, gz4, gz4, gz8, 0);
        __builtin_amdgcn_s_wait_tensorcnt(0);
    }
    __syncthreads();

    const bf16_t* sWw = sW + (size_t)(wave * 16) * H_DIM;  // wave's 16 gate rows
    float* sGw = sG + wave * 128;

    float thr = *thr_p;
    int b  = lane >> 2;            // 0..7
    int hl = lane & 3;             // 0..3
    int h  = nt * 4 + hl;          // 0..1023
    float syn = 0.0f, mem = 0.0f;

    for (int s = 0; s < S_LEN; ++s) {
        const bf16_t* mprev = membuf + (size_t)(s & 1) * (16 * H_DIM);

        v8f c = {};
        for (int k0 = 0; k0 < H_DIM; k0 += 32) {
            v16bf a  = load_a_frag(mprev, 0, H_DIM, k0, lane);   // from L2
            v16bf bb = load_b_frag(sWw, 0, H_DIM, k0, lane);     // from LDS
            c = WMMA_BF16(a, bb, c);
        }
        // Scatter matmul tile (batches 0..7 live in lanes 0..15, vgpr=b).
        if (lane < 16) {
            #pragma unroll
            for (int r = 0; r < 8; ++r) sGw[r * 16 + lane] = c[r];
        }
        __syncthreads();

        // Elementwise update for (b, h): gates at columns hl*4 + {0,1,2,3}.
        size_t t = (size_t)b * S_LEN + s;
        const bf16_t* xgp = xg + t * G4 + (size_t)nt * 16 + hl * 4;
        float gi = sGw[b * 16 + hl * 4 + 0] + from_bf(xgp[0]);
        float gf = sGw[b * 16 + hl * 4 + 1] + from_bf(xgp[1]);
        float gg = sGw[b * 16 + hl * 4 + 2] + from_bf(xgp[2]);
        float go = sGw[b * 16 + hl * 4 + 3] + from_bf(xgp[3]);
        syn = sigmoidf_(gf) * syn + sigmoidf_(gi) * tanhf(gg);
        mem = sigmoidf_(go) * tanhf(syn);
        float spk = (mem - thr) > 0.0f ? 1.0f : 0.0f;

        size_t oidx = t * H_DIM + h;
        syns[oidx] = syn;
        mems[oidx] = mem;
        spks[oidx] = to_bf(spk);
        bf16_t* mnext = membuf + (size_t)((s + 1) & 1) * (16 * H_DIM);
        mnext[b * H_DIM + h] = to_bf(mem);

        // Device-wide barrier (sense = monotonically increasing count).
        __threadfence();
        __syncthreads();
        if (threadIdx.x == 0) {
            __hip_atomic_fetch_add(bar, 1u, __ATOMIC_RELEASE, __HIP_MEMORY_SCOPE_AGENT);
            unsigned tgt = (unsigned)SCAN_WGS * (unsigned)(s + 1);
            while (__hip_atomic_load(bar, __ATOMIC_ACQUIRE, __HIP_MEMORY_SCOPE_AGENT) < tgt) {
                __builtin_amdgcn_s_sleep(1);
            }
        }
        __syncthreads();
        __builtin_amdgcn_fence(__ATOMIC_ACQUIRE, "agent");
    }
}

// ---------------------------------------------------------------------------
// Post GEMMs.  mode 0: out1 = A0.Wᵀ + bias          (bf16 out, K = split)
//              mode 1: out  = DyT([A0|A1].Wᵀ + bias) (fp32 out, K=2048)
// Wave = 4 M-tiles x 4 N-tiles.  N = 1024.  grid (32, 16), block 256.
// ---------------------------------------------------------------------------
__global__ void __launch_bounds__(256) gemm_post(
        const bf16_t* __restrict__ A0, const bf16_t* __restrict__ A1,
        const bf16_t* __restrict__ W, int K, int split,
        const float* __restrict__ bias,
        bf16_t* __restrict__ outb, float* __restrict__ outf,
        const float* __restrict__ gamma, const float* __restrict__ beta,
        const float* __restrict__ alpha_p, int mode) {
    int lane = threadIdx.x & 31;
    int wave = threadIdx.x >> 5;
    int mbase = (blockIdx.x * 8 + wave) * 4;
    int nt0   = blockIdx.y * 4;
    v8f c[16] = {};
    for (int k0 = 0; k0 < K; k0 += 32) {
        const bf16_t* src; int kk;
        if (k0 < split) { src = A0; kk = k0; } else { src = A1; kk = k0 - split; }
        v16bf a[4];
        #pragma unroll
        for (int i = 0; i < 4; ++i)
            a[i] = load_a_frag(src, (mbase + i) * 16, 1024, kk, lane);
        #pragma unroll
        for (int j = 0; j < 4; ++j) {
            v16bf b = load_b_frag(W, (nt0 + j) * 16, K, k0, lane);
            #pragma unroll
            for (int i = 0; i < 4; ++i)
                c[i * 4 + j] = WMMA_BF16(a[i], b, c[i * 4 + j]);
        }
    }
    int n = lane & 15, mhi = (lane >> 4) * 8;
    float alpha = *alpha_p;
    #pragma unroll
    for (int j = 0; j < 4; ++j) {
        int col = (nt0 + j) * 16 + n;
        float bv = bias[col];
        #pragma unroll
        for (int i = 0; i < 4; ++i) {
            int trow = (mbase + i) * 16 + mhi;
            #pragma unroll
            for (int r = 0; r < 8; ++r) {
                float v = c[i * 4 + j][r] + bv;
                if (mode == 0) {
                    outb[(size_t)(trow + r) * 1024 + col] = to_bf(v);
                } else {
                    outf[(size_t)(trow + r) * 1024 + col] =
                        gamma[col] * tanhf(alpha * v) + beta[col];
                }
            }
        }
    }
}

// ---------------------------------------------------------------------------
extern "C" void kernel_launch(void* const* d_in, const int* in_sizes, int n_in,
                              void* d_out, int out_size, void* d_ws, size_t ws_size,
                              hipStream_t stream) {
    (void)in_sizes; (void)n_in; (void)out_size; (void)ws_size;
    const float* x     = (const float*)d_in[0];
    const float* W_ih  = (const float*)d_in[1];
    const float* W_hh  = (const float*)d_in[2];
    const float* b_ih  = (const float*)d_in[3];
    const float* b_hh  = (const float*)d_in[4];
    const float* thr   = (const float*)d_in[5];
    const float* W_lin = (const float*)d_in[6];
    const float* b_lin = (const float*)d_in[7];
    const float* W_mix = (const float*)d_in[8];
    const float* b_mix = (const float*)d_in[9];
    const float* alpha = (const float*)d_in[10];
    const float* gamma = (const float*)d_in[11];
    const float* beta  = (const float*)d_in[12];

    float* outf = (float*)d_out;                       // [B,S,H]
    float* syns = outf + (size_t)NT * H_DIM;           // [B,S,H]
    float* mems = syns + (size_t)NT * H_DIM;           // [B,S,H]

    char* ws = (char*)d_ws;
    size_t off = 0;
    auto alloc = [&](size_t bytes) {
        char* p = ws + off;
        off = (off + bytes + 255) & ~(size_t)255;
        return p;
    };
    bf16_t* xb      = (bf16_t*)alloc((size_t)NT * D_DIM * 2);      // 32 MB
    bf16_t* Wih_p   = (bf16_t*)alloc((size_t)G4 * D_DIM * 2);      //  8 MB
    bf16_t* Whh_p   = (bf16_t*)alloc((size_t)G4 * H_DIM * 2);      //  8 MB
    bf16_t* Wlin_b  = (bf16_t*)alloc((size_t)H_DIM * H_DIM * 2);   //  2 MB
    bf16_t* Wmix_b  = (bf16_t*)alloc((size_t)H_DIM * 2048 * 2);    //  4 MB
    float*  biasg   = (float*)alloc((size_t)G4 * 4);
    bf16_t* xg      = (bf16_t*)alloc((size_t)NT * G4 * 2);         // 128 MB
    bf16_t* spks    = (bf16_t*)alloc((size_t)NT * H_DIM * 2);      // 32 MB
    bf16_t* out1    = (bf16_t*)alloc((size_t)NT * H_DIM * 2);      // 32 MB
    bf16_t* membuf  = (bf16_t*)alloc((size_t)2 * 16 * H_DIM * 2);  // 64 KB
    unsigned int* bar = (unsigned int*)alloc(256);

    const int TPB = 256;
    // --- conversions / init -------------------------------------------------
    conv_f2b<<<(NT * D_DIM + TPB - 1) / TPB, TPB, 0, stream>>>(x, xb, NT * D_DIM);
    conv_w_perm<<<(G4 * D_DIM + TPB - 1) / TPB, TPB, 0, stream>>>(W_ih, Wih_p, D_DIM);
    conv_w_perm<<<(G4 * H_DIM + TPB - 1) / TPB, TPB, 0, stream>>>(W_hh, Whh_p, H_DIM);
    conv_f2b<<<(H_DIM * H_DIM + TPB - 1) / TPB, TPB, 0, stream>>>(W_lin, Wlin_b, H_DIM * H_DIM);
    conv_f2b<<<(H_DIM * 2048 + TPB - 1) / TPB, TPB, 0, stream>>>(W_mix, Wmix_b, H_DIM * 2048);
    conv_bias_perm<<<(G4 + TPB - 1) / TPB, TPB, 0, stream>>>(b_ih, b_hh, biasg);
    init_state<<<(2 * 16 * H_DIM + TPB - 1) / TPB, TPB, 0, stream>>>(
        (unsigned short*)membuf, bar);

    // --- phase 1: xg = x.W_ihᵀ + biases (permuted gate order) ---------------
    gemm_xg<<<dim3(NT / 16 / 32, G4 / 64), TPB, 0, stream>>>(xb, Wih_p, biasg, xg);

    // --- phase 2: persistent recurrence (W_hh in 256KB LDS via TDM) ---------
    size_t scan_lds = (size_t)128 * H_DIM * 2 + 8 * 128 * sizeof(float);
    slstm_scan<<<SCAN_WGS, TPB, scan_lds, stream>>>(
        Whh_p, xg, thr, syns, mems, spks, membuf, bar);

    // --- phase 3: out1 = spks.W_linᵀ + b_lin (bf16) -------------------------
    gemm_post<<<dim3(NT / 16 / 32, H_DIM / 64), TPB, 0, stream>>>(
        spks, spks, Wlin_b, 1024, 1024, b_lin, out1, outf, gamma, beta, alpha, 0);

    // --- phase 4: out = DyT([out1 | x].W_mixᵀ + b_mix) ----------------------
    gemm_post<<<dim3(NT / 16 / 32, H_DIM / 64), TPB, 0, stream>>>(
        out1, xb, Wmix_b, 2048, 1024, b_mix, out1, outf, gamma, beta, alpha, 1);
}
